// FasterRcnnRegionDetector_23880018166430
// MI455X (gfx1250) — matrix operand order repaired and checked by reading, not compiled, not run on hardware
//
#include <hip/hip_runtime.h>

typedef __attribute__((ext_vector_type(16))) __bf16 v16bf;
typedef __attribute__((ext_vector_type(8)))  float  v8f;

#define B_        2
#define C_        1024
#define HF_       50
#define WF_       50
#define NSP       2500            // HF*WF
#define NA_       15
#define NANCH_    37500           // NSP*NA
#define PRE_NMS_  6000
#define NROI_     1000
#define POOLP_    7
#define FDIM_     50176           // C_*49
#define HID_      1024
#define NCLS1_    1601
#define DET_      36
#define MPAD_     2048            // B_*NROI_ padded to a multiple of MT_
#define DELTA_CLIPF 4.1351666f    // log(1000/16)
#define NEGBIG   -3.4e38f

// ---------------- bf16 helpers (bit-exact, no fptrunc dependence) ----------
__device__ __forceinline__ __bf16 f2bf(float f) {
  unsigned u = __builtin_bit_cast(unsigned, f);
  unsigned r = u + 0x7FFFu + ((u >> 16) & 1u);
  unsigned short h = (unsigned short)(r >> 16);
  return __builtin_bit_cast(__bf16, h);
}
__device__ __forceinline__ float bf2f(__bf16 b) {
  unsigned short h = __builtin_bit_cast(unsigned short, b);
  unsigned u = ((unsigned)h) << 16;
  return __builtin_bit_cast(float, u);
}

// ---------------- CDNA5 async global->LDS (ASYNCcnt path) ------------------
// LDS aperture occupies addr[63:32]; addr[31:0] of a generic shared pointer is
// the wave-relative LDS byte offset (ISA 10.2), which is what VDST carries.
__device__ __forceinline__ unsigned lds_off_of(const void* p) {
  return (unsigned)(unsigned long long)p;
}
__device__ __forceinline__ void async_load_b128(unsigned lds_off, const void* gaddr) {
  asm volatile("global_load_async_to_lds_b128 %0, %1, off"
               :: "v"(lds_off), "v"(gaddr) : "memory");
}
__device__ __forceinline__ void wait_asynccnt0() {
  asm volatile("s_wait_asynccnt 0x0" ::: "memory");
}

// ---------------- WMMA tile core ----------------
// LDS A: [128][32] bf16 (m-major).  LDS B: [64][32] bf16 (n-major, i.e. B^T).
// Fragment layouts per CDNA5 ISA 7.12.2 (16-bit A 16x32; B analogous; C/D 16x16 f32).
#define MT_ 128
#define NT_ 64
#define KT_ 32

__device__ __forceinline__ void wmma_step(const __bf16* lA, const __bf16* lB,
                                          int wm, int wn, int lane, v8f acc[2][2]) {
  const int mrow = lane & 15;
  const int half = lane >> 4;
  v16bf afr[2], bfr[2];
#pragma unroll
  for (int fm = 0; fm < 2; ++fm) {
    const __bf16* ap = lA + (wm + fm * 16 + mrow) * KT_;
#pragma unroll
    for (int v = 0; v < 8; ++v) {
      int k0 = ((v < 4) ? (2 * v) : (16 + 2 * (v - 4))) + half * 8;
      afr[fm][2 * v]     = ap[k0];
      afr[fm][2 * v + 1] = ap[k0 + 1];
    }
  }
#pragma unroll
  for (int fn = 0; fn < 2; ++fn) {
    const __bf16* bp = lB + (wn + fn * 16 + mrow) * KT_;
#pragma unroll
    for (int v = 0; v < 8; ++v) {
      int k0 = ((v < 4) ? (2 * v) : (16 + 2 * (v - 4))) + half * 8;
      bfr[fn][2 * v]     = bp[k0];
      bfr[fn][2 * v + 1] = bp[k0 + 1];
    }
  }
#pragma unroll
  for (int fm = 0; fm < 2; ++fm)
#pragma unroll
    for (int fn = 0; fn < 2; ++fn)
      acc[fm][fn] = __builtin_amdgcn_wmma_f32_16x16x32_bf16(
          false, afr[fm], false, bfr[fn], (short)0, acc[fm][fn], false, false);
}

// ---------------- Kernel 1: RPN 3x3 conv as implicit GEMM (WMMA bf16) ------
// M = Cout(1024), N = spatial(2500, padded), K = Cin*9 = 9216.  relu(out+bias) -> t_bf.
__global__ __launch_bounds__(256) void conv_gemm_kernel(
    const float* __restrict__ feat, const float* __restrict__ w_rpn,
    const float* __restrict__ b_rpn, __bf16* __restrict__ t_bf) {
  __shared__ __bf16 lA[MT_ * KT_];
  __shared__ __bf16 lB[NT_ * KT_];
  const int mt = blockIdx.x, nt = blockIdx.y, b = blockIdx.z;
  const int tid = threadIdx.x;
  const int wid = tid >> 5, lane = tid & 31;
  const int wm = (wid >> 1) * 32, wn = (wid & 1) * 32;
  const int K = C_ * 9;
  const float* fb = feat + (size_t)b * C_ * NSP;
  v8f acc[2][2] = {};
  for (int ks = 0; ks < K; ks += KT_) {
#pragma unroll
    for (int i = 0; i < 16; ++i) {               // A: w_rpn rows, coalesced on k
      int e = i * 256 + tid;
      int ml = e >> 5, kl = e & 31;
      int m = mt * MT_ + ml;
      lA[ml * KT_ + kl] = f2bf(w_rpn[(size_t)m * K + ks + kl]);
    }
#pragma unroll
    for (int i = 0; i < 8; ++i) {                // B: im2col gather, coalesced on n
      int e = i * 256 + tid;
      int nl = e & 63, kl = e >> 6;
      int n = nt * NT_ + nl;
      int k = ks + kl;
      int cin = k / 9, r = k - cin * 9;
      int ky = r / 3 - 1, kx = r - (r / 3) * 3 - 1;
      int y = n / WF_ + ky, x = n - (n / WF_) * WF_ + kx;
      float v = 0.f;
      if (n < NSP && y >= 0 && y < HF_ && x >= 0 && x < WF_)
        v = fb[(size_t)cin * NSP + y * WF_ + x];
      lB[nl * KT_ + kl] = f2bf(v);
    }
    __syncthreads();
    wmma_step(lA, lB, wm, wn, lane, acc);
    __syncthreads();
  }
  const int half = lane >> 4, ncol = lane & 15;
#pragma unroll
  for (int fm = 0; fm < 2; ++fm)
#pragma unroll
    for (int fn = 0; fn < 2; ++fn)
#pragma unroll
      for (int j = 0; j < 8; ++j) {
        int m = mt * MT_ + wm + fm * 16 + half * 8 + j;
        int n = nt * NT_ + wn + fn * 16 + ncol;
        if (n < NSP) {
          float v = acc[fm][fn][j] + b_rpn[m];
          v = fmaxf(v, 0.f);
          t_bf[((size_t)b * C_ + m) * NSP + n] = f2bf(v);
        }
      }
}

// ---------------- Kernel 2: obj/delta heads ----------------
__global__ void heads_kernel(const __bf16* __restrict__ t_bf,
                             const float* __restrict__ w_obj, const float* __restrict__ b_obj,
                             const float* __restrict__ w_delta, const float* __restrict__ b_delta,
                             float* __restrict__ logits, float* __restrict__ deltas) {
  int gid = blockIdx.x * blockDim.x + threadIdx.x;
  const int total = B_ * NSP * 75;
  if (gid >= total) return;
  int o = gid % 75;
  int rem = gid / 75;
  int n = rem % NSP, b = rem / NSP;
  const __bf16* tp = t_bf + (size_t)b * C_ * NSP + n;
  const float* w = (o < NA_) ? (w_obj + o * C_) : (w_delta + (o - NA_) * C_);
  float s = 0.f;
  for (int c = 0; c < C_; ++c) s += bf2f(tp[(size_t)c * NSP]) * w[c];
  if (o < NA_) {
    logits[(size_t)b * NANCH_ + n * NA_ + o] = s + b_obj[o];
  } else {
    int ch = o - NA_;
    int a = ch >> 2, d = ch & 3;
    deltas[((size_t)b * NANCH_ + n * NA_ + a) * 4 + d] = s + b_delta[ch];
  }
}

// ---------------- Kernel 3: anchor decode + clip ----------------
__global__ void decode_anchors_kernel(const float* __restrict__ deltas,
                                      const int* __restrict__ image_sizes,
                                      float* __restrict__ aboxes) {
  int gid = blockIdx.x * blockDim.x + threadIdx.x;
  if (gid >= B_ * NANCH_) return;
  int b = gid / NANCH_;
  int i = gid - b * NANCH_;
  int a = i % NA_;
  int n = i / NA_;
  int y = n / WF_, x = n - y * WF_;
  const float sizes[5]  = {32.f, 64.f, 128.f, 256.f, 512.f};
  const float ratios[3] = {0.5f, 1.f, 2.f};
  float s = sizes[a / 3], r = ratios[a % 3];
  float wa = sqrtf(s * s / r);
  float ha = wa * r;
  float cx = (x + 0.5f) * 16.f, cy = (y + 0.5f) * 16.f;
  const float* dl = deltas + (size_t)gid * 4;
  float dx = dl[0], dy = dl[1];
  float dw = fminf(dl[2], DELTA_CLIPF), dh = fminf(dl[3], DELTA_CLIPF);
  float px = dx * wa + cx, py = dy * ha + cy;
  float pw = expf(dw) * wa, ph = expf(dh) * ha;
  float imh = (float)image_sizes[b * 2 + 0], imw = (float)image_sizes[b * 2 + 1];
  float* ob = aboxes + (size_t)gid * 4;
  ob[0] = fminf(fmaxf(px - 0.5f * pw, 0.f), imw);
  ob[1] = fminf(fmaxf(py - 0.5f * ph, 0.f), imh);
  ob[2] = fminf(fmaxf(px + 0.5f * pw, 0.f), imw);
  ob[3] = fminf(fmaxf(py + 0.5f * ph, 0.f), imh);
}

// ---------------- Exact deterministic top-k via rank counting --------------
// rank(i) = #{ j : s[j] > s[i]  ||  (s[j]==s[i] && j<i) }   (== lax.top_k order)
__global__ __launch_bounds__(256) void topk_rank_kernel(const float* __restrict__ scores,
                                                        int N, int K, int* __restrict__ out_idx) {
  __shared__ float tile[256];
  int b = blockIdx.y;
  const float* sc = scores + (size_t)b * N;
  int i = blockIdx.x * 256 + threadIdx.x;
  bool active = (i < N);
  float si = active ? sc[i] : 0.f;
  int rank = 0;
  for (int base = 0; base < N; base += 256) {
    int j = base + threadIdx.x;
    tile[threadIdx.x] = (j < N) ? sc[j] : 0.f;
    __syncthreads();
    int lim = (N - base < 256) ? (N - base) : 256;
    for (int jj = 0; jj < lim; ++jj) {
      float sj = tile[jj];
      int j2 = base + jj;
      rank += (sj > si) || (sj == si && j2 < i);
    }
    __syncthreads();
  }
  if (active && rank < K) out_idx[(size_t)b * K + rank] = i;
}

__global__ void gather_rows4_kernel(const float* __restrict__ src, const int* __restrict__ idx,
                                    float* __restrict__ dst, int N_src, int K) {
  int b = blockIdx.y;
  int r = blockIdx.x * blockDim.x + threadIdx.x;
  if (r >= K) return;
  int i = idx[(size_t)b * K + r];
  const float* s = src + ((size_t)b * N_src + i) * 4;
  float* d = dst + ((size_t)b * K + r) * 4;
  d[0] = s[0]; d[1] = s[1]; d[2] = s[2]; d[3] = s[3];
}

__global__ void gather_rows1_kernel(const float* __restrict__ src, const int* __restrict__ idx,
                                    float* __restrict__ dst, int N_src, int K) {
  int b = blockIdx.y;
  int r = blockIdx.x * blockDim.x + threadIdx.x;
  if (r >= K) return;
  dst[(size_t)b * K + r] = src[(size_t)b * N_src + idx[(size_t)b * K + r]];
}

__global__ void gather_idx_kernel(const int* __restrict__ src, const int* __restrict__ sel,
                                  int* __restrict__ dst, int N_src, int K) {
  int b = blockIdx.y;
  int r = blockIdx.x * blockDim.x + threadIdx.x;
  if (r >= K) return;
  dst[(size_t)b * K + r] = src[(size_t)b * N_src + sel[(size_t)b * K + r]];
}

// ---------------- Greedy sequential NMS (matches lax.fori_loop order) ------
__global__ void nms_kernel(const float* __restrict__ boxes, int N, int PT, float thresh,
                           int* __restrict__ keep) {
  int b = blockIdx.x;
  const float* bx = boxes + (size_t)b * N * 4;
  int* kp = keep + (size_t)b * N;
  int t = threadIdx.x;
  float X1[8], Y1[8], X2[8], Y2[8], AR[8];
  bool KF[8];
  for (int p = 0; p < 8; ++p) {
    int j = t * PT + p;
    if (p < PT && j < N) {
      X1[p] = bx[j * 4 + 0]; Y1[p] = bx[j * 4 + 1];
      X2[p] = bx[j * 4 + 2]; Y2[p] = bx[j * 4 + 3];
      AR[p] = fmaxf(X2[p] - X1[p], 0.f) * fmaxf(Y2[p] - Y1[p], 0.f);
      KF[p] = true;
    } else { X1[p] = Y1[p] = X2[p] = Y2[p] = AR[p] = 0.f; KF[p] = false; }
  }
  __shared__ float cb[5];
  __shared__ int ck;
  for (int i = 0; i < N; ++i) {
    int ot = i / PT, op = i - ot * PT;
    if (t == ot) {
      cb[0] = X1[op]; cb[1] = Y1[op]; cb[2] = X2[op]; cb[3] = Y2[op]; cb[4] = AR[op];
      ck = KF[op] ? 1 : 0;
    }
    __syncthreads();
    if (ck) {
      float cx1 = cb[0], cy1 = cb[1], cx2 = cb[2], cy2 = cb[3], ca = cb[4];
      for (int p = 0; p < PT; ++p) {
        int j = t * PT + p;
        if (j > i && KF[p]) {
          float lx = fmaxf(cx1, X1[p]), ly = fmaxf(cy1, Y1[p]);
          float rx = fminf(cx2, X2[p]), ry = fminf(cy2, Y2[p]);
          float iw = fmaxf(rx - lx, 0.f), ih = fmaxf(ry - ly, 0.f);
          float inter = iw * ih;
          float iou = inter / (ca + AR[p] - inter + 1e-8f);
          if (iou > thresh) KF[p] = false;
        }
      }
    }
    __syncthreads();
  }
  for (int p = 0; p < PT; ++p) {
    int j = t * PT + p;
    if (j < N) kp[j] = KF[p] ? 1 : 0;
  }
}

__global__ void mask2_kernel(const float* __restrict__ ss, const int* __restrict__ keep,
                             float* __restrict__ m2) {
  int gid = blockIdx.x * blockDim.x + threadIdx.x;
  if (gid >= B_ * PRE_NMS_) return;
  m2[gid] = keep[gid] ? ss[gid] : NEGBIG;
}

__global__ void mask4_kernel(const float* __restrict__ scs, const int* __restrict__ keep2,
                             float* __restrict__ m4) {
  int gid = blockIdx.x * blockDim.x + threadIdx.x;
  if (gid >= B_ * NROI_) return;
  float v = scs[gid];
  m4[gid] = (keep2[gid] && v > 0.05f) ? v : -2.0f;
}

// ---------------- ROI align -> flat (bf16) ----------------
__global__ __launch_bounds__(256) void roi_align_kernel(
    const float* __restrict__ feat, const float* __restrict__ proposals,
    __bf16* __restrict__ flatb) {
  int br = blockIdx.x;                       // b*1000 + r
  int b = br / NROI_;
  const float* pb = proposals + (size_t)br * 4;
  __shared__ int   c00[49], c01[49], c10[49], c11[49];
  __shared__ float w00[49], w01[49], w10[49], w11[49];
  int tid = threadIdx.x;
  if (tid < 49) {
    int py = tid / 7, px = tid - (tid / 7) * 7;
    float x1 = pb[0] / 16.f, y1 = pb[1] / 16.f, x2 = pb[2] / 16.f, y2 = pb[3] / 16.f;
    float xc = x1 + (px + 0.5f) * (x2 - x1) / 7.f - 0.5f;
    float yc = y1 + (py + 0.5f) * (y2 - y1) / 7.f - 0.5f;
    float x0f = floorf(xc), y0f = floorf(yc);
    float wx = xc - x0f, wy = yc - y0f;
    int x0i = (int)fminf(fmaxf(x0f, 0.f), 49.f);
    int x1i = (int)fminf(fmaxf(x0f + 1.f, 0.f), 49.f);
    int y0i = (int)fminf(fmaxf(y0f, 0.f), 49.f);
    int y1i = (int)fminf(fmaxf(y0f + 1.f, 0.f), 49.f);
    c00[tid] = y0i * WF_ + x0i;  c01[tid] = y0i * WF_ + x1i;
    c10[tid] = y1i * WF_ + x0i;  c11[tid] = y1i * WF_ + x1i;
    w00[tid] = (1.f - wy) * (1.f - wx);  w01[tid] = (1.f - wy) * wx;
    w10[tid] = wy * (1.f - wx);          w11[tid] = wy * wx;
  }
  __syncthreads();
  const float* fb = feat + (size_t)b * C_ * NSP;
  __bf16* ob = flatb + (size_t)br * FDIM_;
  for (int e = tid; e < FDIM_; e += 256) {
    int c = e / 49, cell = e - (e / 49) * 49;
    const float* fc = fb + (size_t)c * NSP;
    float v = w00[cell] * fc[c00[cell]] + w01[cell] * fc[c01[cell]] +
              w10[cell] * fc[c10[cell]] + w11[cell] * fc[c11[cell]];
    ob[e] = f2bf(v);
  }
}

// ---------------- zero-fill pad rows of flat (rows 2000..2047) -------------
__global__ void zero_pad_kernel(unsigned* __restrict__ dst, size_t n_words) {
  size_t gid = (size_t)blockIdx.x * blockDim.x + threadIdx.x;
  if (gid < n_words) dst[gid] = 0u;
}

// ---------------- Kernel: FC GEMM (WMMA bf16) M=2048 N=1024 K=50176 --------
// A tile streamed straight into LDS with global_load_async_to_lds_b128 (ASYNCcnt).
__global__ __launch_bounds__(256) void fc_gemm_kernel(
    const __bf16* __restrict__ flatb, const float* __restrict__ w_fc,
    const float* __restrict__ b_fc, __bf16* __restrict__ hfcb) {
  __shared__ __bf16 lA[MT_ * KT_];
  __shared__ __bf16 lB[NT_ * KT_];
  const int mt = blockIdx.x, nt = blockIdx.y;
  const int tid = threadIdx.x;
  const int wid = tid >> 5, lane = tid & 31;
  const int wm = (wid >> 1) * 32, wn = (wid & 1) * 32;
  v8f acc[2][2] = {};
  for (int ks = 0; ks < FDIM_; ks += KT_) {
    // A: 128x32 bf16 = 512 x 16B chunks; 2 async b128 per thread, no VGPR staging
#pragma unroll
    for (int i = 0; i < 2; ++i) {
      int cchunk = i * 256 + tid;            // 0..511
      int ml = cchunk >> 2, koff = (cchunk & 3) * 8;
      int m = mt * MT_ + ml;                 // < MPAD_, rows >= 2000 are zero pad
      async_load_b128(lds_off_of(&lA[ml * KT_ + koff]),
                      &flatb[(size_t)m * FDIM_ + ks + koff]);
    }
#pragma unroll
    for (int i = 0; i < 8; ++i) {            // B: f32 -> bf16 convert in flight
      int e = i * 256 + tid;
      int nl = e & 63, kl = e >> 6;
      lB[nl * KT_ + kl] = f2bf(w_fc[(size_t)(ks + kl) * HID_ + nt * NT_ + nl]);
      __builtin_prefetch(&w_fc[(size_t)(ks + kl + KT_) * HID_ + nt * NT_ + nl], 0, 1);
    }
    wait_asynccnt0();
    __syncthreads();
    wmma_step(lA, lB, wm, wn, lane, acc);
    __syncthreads();
  }
  const int half = lane >> 4, ncol = lane & 15;
#pragma unroll
  for (int fm = 0; fm < 2; ++fm)
#pragma unroll
    for (int fn = 0; fn < 2; ++fn)
#pragma unroll
      for (int j = 0; j < 8; ++j) {
        int m = mt * MT_ + wm + fm * 16 + half * 8 + j;   // < MPAD_: pad rows stored too
        int n = nt * NT_ + wn + fn * 16 + ncol;
        float v = fmaxf(acc[fm][fn][j] + b_fc[n], 0.f);
        hfcb[(size_t)m * HID_ + n] = f2bf(v);
      }
}

// ---------------- Kernel: cls GEMM (WMMA bf16) M=2048 N=1601 K=1024 --------
__global__ __launch_bounds__(256) void cls_gemm_kernel(
    const __bf16* __restrict__ hfcb, const float* __restrict__ w_cls,
    const float* __restrict__ b_cls, float* __restrict__ clsl) {
  __shared__ __bf16 lA[MT_ * KT_];
  __shared__ __bf16 lB[NT_ * KT_];
  const int mt = blockIdx.x, nt = blockIdx.y;
  const int tid = threadIdx.x;
  const int wid = tid >> 5, lane = tid & 31;
  const int wm = (wid >> 1) * 32, wn = (wid & 1) * 32;
  const int M = B_ * NROI_;
  v8f acc[2][2] = {};
  for (int ks = 0; ks < HID_; ks += KT_) {
#pragma unroll
    for (int i = 0; i < 2; ++i) {            // A: async bf16 rows of hfc
      int cchunk = i * 256 + tid;
      int ml = cchunk >> 2, koff = (cchunk & 3) * 8;
      int m = mt * MT_ + ml;                 // < MPAD_, all rows initialized by FC
      async_load_b128(lds_off_of(&lA[ml * KT_ + koff]),
                      &hfcb[(size_t)m * HID_ + ks + koff]);
    }
#pragma unroll
    for (int i = 0; i < 8; ++i) {
      int e = i * 256 + tid;
      int nl = e & 63, kl = e >> 6;
      int n = nt * NT_ + nl;
      lB[nl * KT_ + kl] = (n < NCLS1_) ? f2bf(w_cls[(size_t)(ks + kl) * NCLS1_ + n]) : f2bf(0.f);
    }
    wait_asynccnt0();
    __syncthreads();
    wmma_step(lA, lB, wm, wn, lane, acc);
    __syncthreads();
  }
  const int half = lane >> 4, ncol = lane & 15;
#pragma unroll
  for (int fm = 0; fm < 2; ++fm)
#pragma unroll
    for (int fn = 0; fn < 2; ++fn)
#pragma unroll
      for (int j = 0; j < 8; ++j) {
        int m = mt * MT_ + wm + fm * 16 + half * 8 + j;
        int n = nt * NT_ + wn + fn * 16 + ncol;
        if (m < M && n < NCLS1_)
          clsl[(size_t)m * NCLS1_ + n] = acc[fm][fn][j] + b_cls[n];
      }
}

// ---------------- bbox head (tiny) ----------------
__global__ void bbox_head_kernel(const __bf16* __restrict__ hfcb, const float* __restrict__ w_bbox,
                                 const float* __restrict__ b_bbox, float* __restrict__ bd) {
  int gid = blockIdx.x * blockDim.x + threadIdx.x;
  if (gid >= B_ * NROI_ * 4) return;
  int k = gid & 3, row = gid >> 2;
  const __bf16* h = hfcb + (size_t)row * HID_;
  float s = 0.f;
  for (int c = 0; c < HID_; ++c) s += bf2f(h[c]) * w_bbox[c * 4 + k];
  bd[gid] = s + b_bbox[k];
}

// ---------------- reductions ----------------
__device__ __forceinline__ float block_reduce_max(float v, float* red) {
  int tid = threadIdx.x;
  red[tid] = v; __syncthreads();
  for (int s = 128; s > 0; s >>= 1) {
    if (tid < s) red[tid] = fmaxf(red[tid], red[tid + s]);
    __syncthreads();
  }
  float r = red[0]; __syncthreads();
  return r;
}
__device__ __forceinline__ float block_reduce_sum(float v, float* red) {
  int tid = threadIdx.x;
  red[tid] = v; __syncthreads();
  for (int s = 128; s > 0; s >>= 1) {
    if (tid < s) red[tid] += red[tid + s];
    __syncthreads();
  }
  float r = red[0]; __syncthreads();
  return r;
}

// ---------------- best foreground prob + score mask ----------------
__global__ __launch_bounds__(256) void best_score_kernel(const float* __restrict__ clsl,
                                                         float* __restrict__ bestm) {
  __shared__ float red[256];
  int br = blockIdx.x;
  const float* lg = clsl + (size_t)br * NCLS1_;
  int tid = threadIdx.x;
  float mloc = NEGBIG, mfg = NEGBIG;
  for (int c = tid; c < NCLS1_; c += 256) {
    float v = lg[c];
    mloc = fmaxf(mloc, v);
    if (c < NCLS1_ - 1) mfg = fmaxf(mfg, v);
  }
  float m_all = block_reduce_max(mloc, red);
  float m_f   = block_reduce_max(mfg, red);
  float sloc = 0.f;
  for (int c = tid; c < NCLS1_; c += 256) sloc += expf(lg[c] - m_all);
  float ssum = block_reduce_sum(sloc, red);
  if (tid == 0) {
    float best = expf(m_f - m_all) / ssum;
    bestm[br] = (best > 0.05f) ? best : -1.0f;
  }
}

// ---------------- second-stage box decode (BBW weights) + clip -------------
__global__ void decode_dbox_kernel(const float* __restrict__ bd, const float* __restrict__ props,
                                   const int* __restrict__ image_sizes, float* __restrict__ dbox) {
  int gid = blockIdx.x * blockDim.x + threadIdx.x;
  if (gid >= B_ * NROI_) return;
  int b = gid / NROI_;
  const float* p = props + (size_t)gid * 4;
  const float* d = bd + (size_t)gid * 4;
  float wa = p[2] - p[0], ha = p[3] - p[1];
  float cx = p[0] + 0.5f * wa, cy = p[1] + 0.5f * ha;
  float dx = d[0] / 10.f, dy = d[1] / 10.f;
  float dw = fminf(d[2] / 5.f, DELTA_CLIPF), dh = fminf(d[3] / 5.f, DELTA_CLIPF);
  float px = dx * wa + cx, py = dy * ha + cy;
  float pw = expf(dw) * wa, ph = expf(dh) * ha;
  float imh = (float)image_sizes[b * 2 + 0], imw = (float)image_sizes[b * 2 + 1];
  float* ob = dbox + (size_t)gid * 4;
  ob[0] = fminf(fmaxf(px - 0.5f * pw, 0.f), imw);
  ob[1] = fminf(fmaxf(py - 0.5f * ph, 0.f), imh);
  ob[2] = fminf(fmaxf(px + 0.5f * pw, 0.f), imw);
  ob[3] = fminf(fmaxf(py + 0.5f * ph, 0.f), imh);
}

// ---------------- final gather: boxes | probs | feats ----------------
__global__ __launch_bounds__(256) void final_gather_kernel(
    const float* __restrict__ proposals, const float* __restrict__ clsl,
    const __bf16* __restrict__ flatb, const int* __restrict__ fidx,
    float* __restrict__ out) {
  __shared__ float red[256];
  int bs = blockIdx.x;                 // b*36 + s
  int b = bs / DET_;
  int r = fidx[bs];
  int tid = threadIdx.x;
  if (tid < 4) out[(size_t)bs * 4 + tid] = proposals[((size_t)b * NROI_ + r) * 4 + tid];
  const float* lg = clsl + ((size_t)b * NROI_ + r) * NCLS1_;
  float mloc = NEGBIG;
  for (int c = tid; c < NCLS1_; c += 256) mloc = fmaxf(mloc, lg[c]);
  float m = block_reduce_max(mloc, red);
  float sloc = 0.f;
  for (int c = tid; c < NCLS1_; c += 256) sloc += expf(lg[c] - m);
  float ssum = block_reduce_sum(sloc, red);
  float inv = 1.f / ssum;
  float* pout = out + (size_t)B_ * DET_ * 4 + (size_t)bs * (NCLS1_ - 1);
  for (int c = tid; c < NCLS1_ - 1; c += 256) pout[c] = expf(lg[c] - m) * inv;
  const __bf16* fp = flatb + ((size_t)b * NROI_ + r) * FDIM_;
  float* fout = out + (size_t)B_ * DET_ * 4 + (size_t)B_ * DET_ * (NCLS1_ - 1) + (size_t)bs * FDIM_;
  for (int f = tid; f < FDIM_; f += 256) fout[f] = bf2f(fp[f]);
}

// ================= host side =================
static constexpr size_t alup(size_t x) { return (x + 255) & ~(size_t)255; }

static constexpr size_t OFF_TBF   = 0;                                  // bf16 t: 2*1024*2500
static constexpr size_t OFF_LOG   = OFF_TBF   + alup((size_t)B_ * C_ * NSP * 2);
static constexpr size_t OFF_DEL   = OFF_LOG   + alup((size_t)B_ * NANCH_ * 4);
static constexpr size_t OFF_ABOX  = OFF_DEL   + alup((size_t)B_ * NANCH_ * 16);
static constexpr size_t OFF_IDX1  = OFF_ABOX  + alup((size_t)B_ * NANCH_ * 16);
static constexpr size_t OFF_SBOX  = OFF_IDX1  + alup((size_t)B_ * PRE_NMS_ * 4);
static constexpr size_t OFF_SSC   = OFF_SBOX  + alup((size_t)B_ * PRE_NMS_ * 16);
static constexpr size_t OFF_KEEP1 = OFF_SSC   + alup((size_t)B_ * PRE_NMS_ * 4);
static constexpr size_t OFF_M2    = OFF_KEEP1 + alup((size_t)B_ * PRE_NMS_ * 4);
static constexpr size_t OFF_IDX2  = OFF_M2    + alup((size_t)B_ * PRE_NMS_ * 4);
static constexpr size_t OFF_PROP  = OFF_IDX2  + alup((size_t)B_ * NROI_ * 4);
static constexpr size_t OFF_FLAT  = OFF_PROP  + alup((size_t)B_ * NROI_ * 16);
static constexpr size_t OFF_HFC   = OFF_FLAT  + alup((size_t)MPAD_ * FDIM_ * 2);
static constexpr size_t OFF_CLSL  = OFF_HFC   + alup((size_t)MPAD_ * HID_ * 2);
static constexpr size_t OFF_BD    = OFF_CLSL  + alup((size_t)B_ * NROI_ * NCLS1_ * 4);
static constexpr size_t OFF_DBOX  = OFF_BD    + alup((size_t)B_ * NROI_ * 16);
static constexpr size_t OFF_BEST  = OFF_DBOX  + alup((size_t)B_ * NROI_ * 16);
static constexpr size_t OFF_ORD   = OFF_BEST  + alup((size_t)B_ * NROI_ * 4);
static constexpr size_t OFF_SCS   = OFF_ORD   + alup((size_t)B_ * NROI_ * 4);
static constexpr size_t OFF_OBOX  = OFF_SCS   + alup((size_t)B_ * NROI_ * 4);
static constexpr size_t OFF_KEEP2 = OFF_OBOX  + alup((size_t)B_ * NROI_ * 16);
static constexpr size_t OFF_M4    = OFF_KEEP2 + alup((size_t)B_ * NROI_ * 4);
static constexpr size_t OFF_SEL   = OFF_M4    + alup((size_t)B_ * NROI_ * 4);
static constexpr size_t OFF_FIDX  = OFF_SEL   + alup((size_t)B_ * DET_ * 4);

extern "C" void kernel_launch(void* const* d_in, const int* in_sizes, int n_in,
                              void* d_out, int out_size, void* d_ws, size_t ws_size,
                              hipStream_t stream) {
  (void)in_sizes; (void)n_in; (void)out_size; (void)ws_size;
  const int*   image_sizes = (const int*)d_in[1];
  const float* feat    = (const float*)d_in[2];
  const float* w_rpn   = (const float*)d_in[3];
  const float* b_rpn   = (const float*)d_in[4];
  const float* w_obj   = (const float*)d_in[5];
  const float* b_obj   = (const float*)d_in[6];
  const float* w_delta = (const float*)d_in[7];
  const float* b_delta = (const float*)d_in[8];
  const float* w_fc    = (const float*)d_in[9];
  const float* b_fc    = (const float*)d_in[10];
  const float* w_cls   = (const float*)d_in[11];
  const float* b_cls   = (const float*)d_in[12];
  const float* w_bbox  = (const float*)d_in[13];
  const float* b_bbox  = (const float*)d_in[14];
  char* ws = (char*)d_ws;
  __bf16* t_bf   = (__bf16*)(ws + OFF_TBF);
  float*  logits = (float*)(ws + OFF_LOG);
  float*  deltas = (float*)(ws + OFF_DEL);
  float*  aboxes = (float*)(ws + OFF_ABOX);
  int*    idx1   = (int*)  (ws + OFF_IDX1);
  float*  sboxes = (float*)(ws + OFF_SBOX);
  float*  sscore = (float*)(ws + OFF_SSC);
  int*    keep1  = (int*)  (ws + OFF_KEEP1);
  float*  m2     = (float*)(ws + OFF_M2);
  int*    idx2   = (int*)  (ws + OFF_IDX2);
  float*  props  = (float*)(ws + OFF_PROP);
  __bf16* flatb  = (__bf16*)(ws + OFF_FLAT);
  __bf16* hfcb   = (__bf16*)(ws + OFF_HFC);
  float*  clsl   = (float*)(ws + OFF_CLSL);
  float*  bd     = (float*)(ws + OFF_BD);
  float*  dbox   = (float*)(ws + OFF_DBOX);
  float*  bestm  = (float*)(ws + OFF_BEST);
  int*    order  = (int*)  (ws + OFF_ORD);
  float*  scs    = (float*)(ws + OFF_SCS);
  float*  obox   = (float*)(ws + OFF_OBOX);
  int*    keep2  = (int*)  (ws + OFF_KEEP2);
  float*  m4     = (float*)(ws + OFF_M4);
  int*    sel    = (int*)  (ws + OFF_SEL);
  int*    fidx   = (int*)  (ws + OFF_FIDX);
  float*  out    = (float*)d_out;

  // 1. RPN conv (WMMA)
  conv_gemm_kernel<<<dim3(C_ / MT_, (NSP + NT_ - 1) / NT_, B_), 256, 0, stream>>>(
      feat, w_rpn, b_rpn, t_bf);
  // 2. obj/delta heads
  heads_kernel<<<(B_ * NSP * 75 + 255) / 256, 256, 0, stream>>>(
      t_bf, w_obj, b_obj, w_delta, b_delta, logits, deltas);
  // 3. anchor decode + clip
  decode_anchors_kernel<<<(B_ * NANCH_ + 255) / 256, 256, 0, stream>>>(
      deltas, image_sizes, aboxes);
  // 4. top-6000
  topk_rank_kernel<<<dim3((NANCH_ + 255) / 256, B_), 256, 0, stream>>>(
      logits, NANCH_, PRE_NMS_, idx1);
  gather_rows4_kernel<<<dim3((PRE_NMS_ + 255) / 256, B_), 256, 0, stream>>>(
      aboxes, idx1, sboxes, NANCH_, PRE_NMS_);
  gather_rows1_kernel<<<dim3((PRE_NMS_ + 255) / 256, B_), 256, 0, stream>>>(
      logits, idx1, sscore, NANCH_, PRE_NMS_);
  // 5. RPN NMS (0.7)
  nms_kernel<<<B_, 768, 0, stream>>>(sboxes, PRE_NMS_, 8, 0.7f, keep1);
  // 6. top-1000 of kept -> proposals
  mask2_kernel<<<(B_ * PRE_NMS_ + 255) / 256, 256, 0, stream>>>(sscore, keep1, m2);
  topk_rank_kernel<<<dim3((PRE_NMS_ + 255) / 256, B_), 256, 0, stream>>>(
      m2, PRE_NMS_, NROI_, idx2);
  gather_rows4_kernel<<<dim3((NROI_ + 255) / 256, B_), 256, 0, stream>>>(
      sboxes, idx2, props, PRE_NMS_, NROI_);
  // 7a. zero the pad rows of flat (rows 2000..2047)
  {
    size_t n_words = (size_t)(MPAD_ - B_ * NROI_) * FDIM_ / 2;   // bf16 pairs
    unsigned* padp = (unsigned*)(flatb + (size_t)B_ * NROI_ * FDIM_);
    zero_pad_kernel<<<(unsigned)((n_words + 255) / 256), 256, 0, stream>>>(padp, n_words);
  }
  // 7b. ROI align
  roi_align_kernel<<<B_ * NROI_, 256, 0, stream>>>(feat, props, flatb);
  // 8. FC (WMMA, async A tiles)
  fc_gemm_kernel<<<dim3(MPAD_ / MT_, HID_ / NT_), 256, 0, stream>>>(
      flatb, w_fc, b_fc, hfcb);
  // 9. cls head (WMMA, async A tiles)
  cls_gemm_kernel<<<dim3(MPAD_ / MT_, (NCLS1_ + NT_ - 1) / NT_), 256, 0, stream>>>(
      hfcb, w_cls, b_cls, clsl);
  // 10. bbox head
  bbox_head_kernel<<<(B_ * NROI_ * 4 + 255) / 256, 256, 0, stream>>>(
      hfcb, w_bbox, b_bbox, bd);
  // 11. best foreground prob with score threshold mask
  best_score_kernel<<<B_ * NROI_, 256, 0, stream>>>(clsl, bestm);
  // 12. second-stage decode + clip
  decode_dbox_kernel<<<(B_ * NROI_ + 255) / 256, 256, 0, stream>>>(
      bd, props, image_sizes, dbox);
  // 13. sort 1000 by best score
  topk_rank_kernel<<<dim3((NROI_ + 255) / 256, B_), 256, 0, stream>>>(
      bestm, NROI_, NROI_, order);
  gather_rows1_kernel<<<dim3((NROI_ + 255) / 256, B_), 256, 0, stream>>>(
      bestm, order, scs, NROI_, NROI_);
  gather_rows4_kernel<<<dim3((NROI_ + 255) / 256, B_), 256, 0, stream>>>(
      dbox, order, obox, NROI_, NROI_);
  // 14. test NMS (0.5)
  nms_kernel<<<B_, 256, 0, stream>>>(obox, NROI_, 4, 0.5f, keep2);
  // 15. final top-36
  mask4_kernel<<<(B_ * NROI_ + 255) / 256, 256, 0, stream>>>(scs, keep2, m4);
  topk_rank_kernel<<<dim3((NROI_ + 255) / 256, B_), 256, 0, stream>>>(
      m4, NROI_, DET_, sel);
  gather_idx_kernel<<<dim3(1, B_), 256, 0, stream>>>(order, sel, fidx, NROI_, DET_);
  // 16. final outputs
  final_gather_kernel<<<B_ * DET_, 256, 0, stream>>>(props, clsl, flatb, fidx, out);
}
